// MultiheadAttention_42958262895123
// MI455X (gfx1250) — matrix operand order repaired and checked by reading, not compiled, not run on hardware
//
#include <hip/hip_runtime.h>

typedef _Float16 half_t;
typedef __attribute__((ext_vector_type(8)))  _Float16 v8h_t;
typedef __attribute__((ext_vector_type(16))) _Float16 v16h_t;
typedef __attribute__((ext_vector_type(8)))  float    v8f_t;

namespace {
constexpr int T_ = 1024, B_ = 2, D_ = 1024;
constexpr int E_ = 12, TK_ = 4, ED_ = 256, HD_ = 64, H_ = 4, MAXP_ = 64;
constexpr int N_ = T_ * B_;          // 2048 tokens
constexpr int P_ = 2 * MAXP_ + 1;    // 129
constexpr int PP_ = 144;             // padded to multiple of 16
constexpr int EED_ = E_ * ED_;       // 3072
constexpr float QK_SCALE = 0.35355339059327373f;  // 64^-0.25
}

#define WMMA_F16(a, b, c) \
  __builtin_amdgcn_wmma_f32_16x16x32_f16(false, (a), false, (b), (short)0, (c), false, false)

// ---------- WMMA fragment loaders (wave32 layouts per CDNA5 ISA 7.12.2) ----
// A (16x32 f16): lane L -> row M=L&15; a[i]   = A[M][k0 + hi*8 + i]      i<8
//                                      a[8+i] = A[M][k0 + 16 + hi*8 + i]
__device__ __forceinline__ v16h_t frag_a(const half_t* base, int lda, int lane, int k0) {
  const int m = lane & 15, hi = lane >> 4;
  const half_t* p = base + (size_t)m * lda + k0 + hi * 8;
  v8h_t lo = *(const v8h_t*)(p);
  v8h_t hh = *(const v8h_t*)(p + 16);
  v16h_t r;
#pragma unroll
  for (int i = 0; i < 8; ++i) { r[i] = lo[i]; r[i + 8] = hh[i]; }
  return r;
}
// B (32x16 f16) from column-major store Bt[N][K]: lane L -> col N=L&15;
//   b[i] = B[k0 + hi*8 + i][N], b[8+i] = B[k0 + 16 + hi*8 + i][N]  (matches A)
__device__ __forceinline__ v16h_t frag_b(const half_t* bt, int ldb, int lane, int col0, int k0) {
  const int n = lane & 15, hi = lane >> 4;
  const half_t* p = bt + (size_t)(col0 + n) * ldb + k0 + hi * 8;
  v8h_t b0 = *(const v8h_t*)(p);
  v8h_t b1 = *(const v8h_t*)(p + 16);
  v16h_t r;
#pragma unroll
  for (int i = 0; i < 8; ++i) { r[i] = b0[i]; r[i + 8] = b1[i]; }
  return r;
}
// C/D f32 16x16: lane L -> col N=L&15, reg r -> row r + 8*(L>>4)

// Software-pipelined 16x64 wave tile: C[16, col0..col0+63] += A[16,K] * B[K,64]
// Double-buffers A/B fragments so global loads of step k+1 overlap the four
// WMMAs of step k (WMMA issues like TRANS ops and co-executes with VMEM/VALU).
__device__ __forceinline__ void gemm_wave_tile(const half_t* __restrict__ arow, int lda,
                                               const half_t* __restrict__ bt, int ldb,
                                               int col0, int K, int lane, v8f_t acc[4]) {
  v16h_t a = frag_a(arow, lda, lane, 0);
  v16h_t b0 = frag_b(bt, ldb, lane, col0 +  0, 0);
  v16h_t b1 = frag_b(bt, ldb, lane, col0 + 16, 0);
  v16h_t b2 = frag_b(bt, ldb, lane, col0 + 32, 0);
  v16h_t b3 = frag_b(bt, ldb, lane, col0 + 48, 0);
  for (int k0 = 32; k0 < K; k0 += 32) {
    // issue next-step loads first so they retire behind this step's WMMAs
    v16h_t na = frag_a(arow, lda, lane, k0);
    v16h_t n0 = frag_b(bt, ldb, lane, col0 +  0, k0);
    v16h_t n1 = frag_b(bt, ldb, lane, col0 + 16, k0);
    v16h_t n2 = frag_b(bt, ldb, lane, col0 + 32, k0);
    v16h_t n3 = frag_b(bt, ldb, lane, col0 + 48, k0);
    __builtin_prefetch(arow + ((size_t)(lane & 15) * lda) + k0 + 32, 0, 3);
    acc[0] = WMMA_F16(a, b0, acc[0]);
    acc[1] = WMMA_F16(a, b1, acc[1]);
    acc[2] = WMMA_F16(a, b2, acc[2]);
    acc[3] = WMMA_F16(a, b3, acc[3]);
    a = na; b0 = n0; b1 = n1; b2 = n2; b3 = n3;
  }
  acc[0] = WMMA_F16(a, b0, acc[0]);
  acc[1] = WMMA_F16(a, b1, acc[1]);
  acc[2] = WMMA_F16(a, b2, acc[2]);
  acc[3] = WMMA_F16(a, b3, acc[3]);
}

// ---------------------------- prep kernels ---------------------------------
__global__ void cvt_f32_f16(const float* __restrict__ src, half_t* __restrict__ dst, int count) {
  for (int i = blockIdx.x * blockDim.x + threadIdx.x; i < count; i += gridDim.x * blockDim.x)
    dst[i] = (half_t)src[i];
}

// dst[b][c][r] = src[b][r][c] (f32 -> f16), batched transpose-convert
__global__ void tr_cvt(const float* __restrict__ src, half_t* __restrict__ dst,
                       int R, int C, int batch) {
  size_t total = (size_t)batch * R * C;
  for (size_t i = blockIdx.x * (size_t)blockDim.x + threadIdx.x; i < total;
       i += (size_t)gridDim.x * blockDim.x) {
    size_t bb = i / ((size_t)R * C);
    size_t rem = i - bb * (size_t)R * C;
    int r = (int)(rem / C), c = (int)(rem % C);
    dst[bb * (size_t)R * C + (size_t)c * R + r] = (half_t)src[i];
  }
}

// rpe (H, HD, P) -> rpet (H, PP, HD) transposed + zero-padded
__global__ void rpe_pad(const float* __restrict__ rpe, half_t* __restrict__ rpet) {
  int total = H_ * PP_ * HD_;
  for (int i = blockIdx.x * blockDim.x + threadIdx.x; i < total; i += gridDim.x * blockDim.x) {
    int hd = i % HD_;
    int p  = (i / HD_) % PP_;
    int h  = i / (HD_ * PP_);
    float v = (p < P_) ? rpe[((size_t)h * HD_ + hd) * P_ + p] : 0.0f;
    rpet[i] = (half_t)v;
  }
}

// vectorized fill (count must be a multiple of 8 halves = 16B)
__global__ void zero_f16_v(half_t* __restrict__ p, int count8) {
  float4* q = (float4*)p;
  float4 z; z.x = z.y = z.z = z.w = 0.0f;
  for (int i = blockIdx.x * blockDim.x + threadIdx.x; i < count8; i += gridDim.x * blockDim.x)
    q[i] = z;
}

// --------------------------- gating + top-k --------------------------------
__global__ __launch_bounds__(128) void gate_topk(const float* __restrict__ x,
                                                 const float* __restrict__ wg,
                                                 float* __restrict__ gates,
                                                 int* __restrict__ tidx) {
  int n = blockIdx.x;
  int tid = threadIdx.x;
  float s[E_];
#pragma unroll
  for (int e = 0; e < E_; ++e) s[e] = 0.0f;
  const float* xr = x + (size_t)n * D_;
  for (int d = tid; d < D_; d += 128) {
    float xv = xr[d];
#pragma unroll
    for (int e = 0; e < E_; ++e) s[e] += xv * wg[(size_t)d * E_ + e];
  }
  __shared__ float red[E_][128];
#pragma unroll
  for (int e = 0; e < E_; ++e) red[e][tid] = s[e];
  __syncthreads();
  for (int off = 64; off > 0; off >>= 1) {
    if (tid < off)
#pragma unroll
      for (int e = 0; e < E_; ++e) red[e][tid] += red[e][tid + off];
    __syncthreads();
  }
  if (tid == 0) {
    float prob[E_];
    float mx = -1e30f;
#pragma unroll
    for (int e = 0; e < E_; ++e) mx = fmaxf(mx, red[e][0]);
    float sum = 0.0f;
#pragma unroll
    for (int e = 0; e < E_; ++e) { prob[e] = __expf(red[e][0] - mx); sum += prob[e]; }
    bool used[E_];
#pragma unroll
    for (int e = 0; e < E_; ++e) used[e] = false;
    for (int k = 0; k < TK_; ++k) {
      int best = 0; float bv = -1.0f;
#pragma unroll
      for (int e = 0; e < E_; ++e)
        if (!used[e] && prob[e] > bv) { bv = prob[e]; best = e; }
      used[best] = true;
      gates[n * TK_ + k] = bv / sum;
      tidx[n * TK_ + k] = best;
    }
  }
}

// --------------------------- WMMA GEMM kernels -----------------------------
// all_map: (2048 x 1024) @ wmap_t-as-(3072 x 1024) -> f16 (2048 x 3072)
__global__ __launch_bounds__(128) void gemm_allmap(const half_t* __restrict__ A,
                                                   const half_t* __restrict__ Bt,
                                                   half_t* __restrict__ C) {
  int lane = threadIdx.x & 31, wave = threadIdx.x >> 5;
  int row0 = (blockIdx.y * 4 + wave) * 16;
  int col0 = blockIdx.x * 64;
  v8f_t acc[4] = {};
  gemm_wave_tile(A + (size_t)row0 * D_, D_, Bt, D_, col0, D_, lane, acc);
  int nn = lane & 15, hi = lane >> 4;
#pragma unroll
  for (int j = 0; j < 4; ++j)
#pragma unroll
    for (int r = 0; r < 8; ++r)
      C[(size_t)(row0 + r + 8 * hi) * EED_ + col0 + 16 * j + nn] = (half_t)acc[j][r];
}

// K/V projection: (2048 x 1024) @ (256 x 1024)^T; mode 0 -> K scaled [b][h][t][hd],
// mode 1 -> V transposed [b][h][hd][t]
__global__ __launch_bounds__(128) void gemm_proj(const half_t* __restrict__ A,
                                                 const half_t* __restrict__ Bt,
                                                 half_t* __restrict__ dst, int mode) {
  int lane = threadIdx.x & 31, wave = threadIdx.x >> 5;
  int row0 = (blockIdx.y * 4 + wave) * 16;
  int col0 = blockIdx.x * 64;
  v8f_t acc[4] = {};
  gemm_wave_tile(A + (size_t)row0 * D_, D_, Bt, D_, col0, D_, lane, acc);
  int nn = lane & 15, hi = lane >> 4;
#pragma unroll
  for (int j = 0; j < 4; ++j)
#pragma unroll
    for (int r = 0; r < 8; ++r) {
      int row = row0 + r + 8 * hi;     // token n = t*B + b
      int col = col0 + 16 * j + nn;    // ED column = h*64 + hd
      int t = row / B_, b = row % B_;
      int h = col / HD_, hd = col % HD_;
      float v = acc[j][r];
      if (mode == 0)
        dst[(((size_t)b * H_ + h) * T_ + t) * HD_ + hd] = (half_t)(v * QK_SCALE);
      else
        dst[(((size_t)b * H_ + h) * HD_ + hd) * T_ + t] = (half_t)v;
    }
}

// final: z (2048 x 3072) @ wr_t-as-(1024 x 3072) -> f32 out (2048 x 1024)
__global__ __launch_bounds__(128) void gemm_out(const half_t* __restrict__ A,
                                                const half_t* __restrict__ Bt,
                                                float* __restrict__ C) {
  int lane = threadIdx.x & 31, wave = threadIdx.x >> 5;
  int row0 = (blockIdx.y * 4 + wave) * 16;
  int col0 = blockIdx.x * 64;
  v8f_t acc[4] = {};
  gemm_wave_tile(A + (size_t)row0 * EED_, EED_, Bt, EED_, col0, EED_, lane, acc);
  int nn = lane & 15, hi = lane >> 4;
#pragma unroll
  for (int j = 0; j < 4; ++j)
#pragma unroll
    for (int r = 0; r < 8; ++r)
      C[(size_t)(row0 + r + 8 * hi) * D_ + col0 + 16 * j + nn] = acc[j][r];
}

// q gather: qh[b][kk][h][t][hd] = QK_SCALE * allmap[n][idx[n][kk]*256 + h*64+hd]
__global__ void q_gather(const half_t* __restrict__ allmap, const int* __restrict__ tidx,
                         half_t* __restrict__ qh) {
  int total = B_ * TK_ * H_ * T_ * HD_;
  for (int u = blockIdx.x * blockDim.x + threadIdx.x; u < total; u += gridDim.x * blockDim.x) {
    int hd = u % HD_;
    int t  = (u / HD_) % T_;
    int h  = (u / (HD_ * T_)) % H_;
    int kk = (u / (HD_ * T_ * H_)) % TK_;
    int b  = u / (HD_ * T_ * H_ * TK_);
    int n = t * B_ + b;
    int e = tidx[n * TK_ + kk];
    float v = (float)allmap[(size_t)n * EED_ + e * ED_ + h * HD_ + hd] * QK_SCALE;
    qh[u] = (half_t)v;
  }
}

// ----------------------- flash attention (1 wave/block) --------------------
__global__ __launch_bounds__(32) void attn_kernel(const half_t* __restrict__ qh,   // [B][TK][H][T][HD]
                                                  const half_t* __restrict__ khp,  // [B][H][T][HD] scaled
                                                  const half_t* __restrict__ vtb,  // [B][H][HD][T]
                                                  const half_t* __restrict__ rpet, // [H][PP][HD]
                                                  const float* __restrict__ gates, // [N][TK]
                                                  const int* __restrict__ tidx,    // [N][TK]
                                                  half_t* __restrict__ z) {        // [N][E][ED]
  __shared__ float Rl[16][PP_];                     // rel bias, this wave's 16 rows
  __shared__ __align__(16) half_t Pl[16][32];       // P staging (C-layout -> A-layout)

  int lane = threadIdx.x;
  int nn = lane & 15, hi = lane >> 4;
  int i0 = blockIdx.x * 16;
  int y = blockIdx.y;                 // = b*TK*H + kk*H + h
  int h = y % H_;
  int kk = (y / H_) % TK_;
  int b = y / (H_ * TK_);

  const half_t* qrow = qh + ((((size_t)b * TK_ + kk) * H_ + h) * T_ + i0) * HD_;
  const half_t* kbh  = khp + (((size_t)b * H_ + h) * T_) * HD_;
  const half_t* vbh  = vtb + (((size_t)b * H_ + h) * HD_) * T_;
  const half_t* rph  = rpet + (size_t)h * PP_ * HD_;

  v16h_t aq0 = frag_a(qrow, HD_, lane, 0);
  v16h_t aq1 = frag_a(qrow, HD_, lane, 32);

  // R = Q @ rpe_t : 16 x 144, into LDS
#pragma unroll
  for (int pt = 0; pt < PP_ / 16; ++pt) {
    v8f_t rc = {};
    rc = WMMA_F16(aq0, frag_b(rph, HD_, lane, pt * 16, 0), rc);
    rc = WMMA_F16(aq1, frag_b(rph, HD_, lane, pt * 16, 32), rc);
#pragma unroll
    for (int r = 0; r < 8; ++r) Rl[r + 8 * hi][pt * 16 + nn] = rc[r];
  }

  float m_i[8], l_i[8];
#pragma unroll
  for (int r = 0; r < 8; ++r) { m_i[r] = -1e30f; l_i[r] = 0.0f; }
  v8f_t o0 = {}, o1 = {}, o2 = {}, o3 = {};

  for (int j0 = 0; j0 < T_; j0 += 32) {
    // scores: two 16x16 tiles (columns j0.., j0+16..), K-dim = HD = 64
    v8f_t s0 = {}, s1 = {};
    s0 = WMMA_F16(aq0, frag_b(kbh, HD_, lane, j0, 0), s0);
    s0 = WMMA_F16(aq1, frag_b(kbh, HD_, lane, j0, 32), s0);
    s1 = WMMA_F16(aq0, frag_b(kbh, HD_, lane, j0 + 16, 0), s1);
    s1 = WMMA_F16(aq1, frag_b(kbh, HD_, lane, j0 + 16, 32), s1);

    // issue V-tile loads now so they overlap the softmax VALU chain below
    v16h_t bv0 = frag_b(vbh, T_, lane, 0, j0);
    v16h_t bv1 = frag_b(vbh, T_, lane, 16, j0);
    v16h_t bv2 = frag_b(vbh, T_, lane, 32, j0);
    v16h_t bv3 = frag_b(vbh, T_, lane, 48, j0);

    // add rel-pos bias gathered by clip(j-i)
#pragma unroll
    for (int r = 0; r < 8; ++r) {
      int i = i0 + r + 8 * hi;
      int ja = j0 + nn, jb = ja + 16;
      int ga = ja - i; ga = ga < (1 - MAXP_) ? (1 - MAXP_) : (ga > (MAXP_ - 1) ? (MAXP_ - 1) : ga);
      int gb = jb - i; gb = gb < (1 - MAXP_) ? (1 - MAXP_) : (gb > (MAXP_ - 1) ? (MAXP_ - 1) : gb);
      s0[r] += Rl[r + 8 * hi][ga + MAXP_];
      s1[r] += Rl[r + 8 * hi][gb + MAXP_];
    }

    // online softmax (rows live across 16 lanes of each half-wave)
#pragma unroll
    for (int r = 0; r < 8; ++r) {
      float mx = fmaxf(s0[r], s1[r]);
#pragma unroll
      for (int off = 8; off >= 1; off >>= 1) mx = fmaxf(mx, __shfl_xor(mx, off, 32));
      float mnew = fmaxf(m_i[r], mx);
      float sc = __expf(m_i[r] - mnew);
      float p0 = __expf(s0[r] - mnew);
      float p1 = __expf(s1[r] - mnew);
      s0[r] = p0; s1[r] = p1;
      float rs = p0 + p1;
#pragma unroll
      for (int off = 8; off >= 1; off >>= 1) rs += __shfl_xor(rs, off, 32);
      l_i[r] = l_i[r] * sc + rs;
      m_i[r] = mnew;
      o0[r] *= sc; o1[r] *= sc; o2[r] *= sc; o3[r] *= sc;
    }

    // P (C-layout) -> LDS -> A-layout
#pragma unroll
    for (int r = 0; r < 8; ++r) {
      Pl[r + 8 * hi][nn] = (half_t)s0[r];
      Pl[r + 8 * hi][16 + nn] = (half_t)s1[r];
    }
    v16h_t ap = frag_a(&Pl[0][0], 32, lane, 0);

    // O += P @ V   (V transposed store makes B contiguous)
    o0 = WMMA_F16(ap, bv0, o0);
    o1 = WMMA_F16(ap, bv1, o1);
    o2 = WMMA_F16(ap, bv2, o2);
    o3 = WMMA_F16(ap, bv3, o3);
  }

  // normalize, apply gate, scatter into expert slot of z (distinct e per kk -> no atomics)
#pragma unroll
  for (int r = 0; r < 8; ++r) {
    int i = i0 + r + 8 * hi;
    int ntok = i * B_ + b;
    float g = gates[ntok * TK_ + kk];
    int e = tidx[ntok * TK_ + kk];
    float inv = g / l_i[r];
    half_t* zp = z + ((size_t)ntok * E_ + e) * ED_ + h * HD_;
    zp[nn]      = (half_t)(o0[r] * inv);
    zp[16 + nn] = (half_t)(o1[r] * inv);
    zp[32 + nn] = (half_t)(o2[r] * inv);
    zp[48 + nn] = (half_t)(o3[r] * inv);
  }
}

// ------------------------------ orchestration ------------------------------
extern "C" void kernel_launch(void* const* d_in, const int* in_sizes, int n_in,
                              void* d_out, int out_size, void* d_ws, size_t ws_size,
                              hipStream_t stream) {
  const float* in_q   = (const float*)d_in[0];
  const float* in_k   = (const float*)d_in[1];
  const float* in_v   = (const float*)d_in[2];
  const float* in_wg  = (const float*)d_in[3];
  const float* in_wm  = (const float*)d_in[4];
  const float* in_wr  = (const float*)d_in[5];
  const float* in_wk  = (const float*)d_in[6];
  const float* in_wv  = (const float*)d_in[7];
  const float* in_rpe = (const float*)d_in[8];
  float* out = (float*)d_out;

  char* ws = (char*)d_ws;
  size_t off = 0;
  auto alloc = [&](size_t bytes) -> void* {
    off = (off + 255) & ~(size_t)255;
    void* p = ws + off;
    off += bytes;
    return p;
  };

  half_t* xh      = (half_t*)alloc((size_t)N_ * D_ * 2);
  half_t* kh      = (half_t*)alloc((size_t)N_ * D_ * 2);
  half_t* vh      = (half_t*)alloc((size_t)N_ * D_ * 2);
  half_t* wmap_t  = (half_t*)alloc((size_t)E_ * ED_ * D_ * 2);
  half_t* wk_t    = (half_t*)alloc((size_t)ED_ * D_ * 2);
  half_t* wv_t    = (half_t*)alloc((size_t)ED_ * D_ * 2);
  half_t* wr_t    = (half_t*)alloc((size_t)D_ * EED_ * 2);
  half_t* rpet    = (half_t*)alloc((size_t)H_ * PP_ * HD_ * 2);
  float*  gates   = (float*)alloc((size_t)N_ * TK_ * 4);
  int*    tidx    = (int*)alloc((size_t)N_ * TK_ * 4);
  half_t* allmap  = (half_t*)alloc((size_t)N_ * EED_ * 2);
  half_t* qh      = (half_t*)alloc((size_t)B_ * TK_ * H_ * T_ * HD_ * 2);
  half_t* khp     = (half_t*)alloc((size_t)B_ * H_ * T_ * HD_ * 2);
  half_t* vtb     = (half_t*)alloc((size_t)B_ * H_ * T_ * HD_ * 2);
  half_t* zbuf    = (half_t*)alloc((size_t)N_ * EED_ * 2);

  // 1) f32 -> f16 activations
  cvt_f32_f16<<<2048, 256, 0, stream>>>(in_q, xh, N_ * D_);
  cvt_f32_f16<<<2048, 256, 0, stream>>>(in_k, kh, N_ * D_);
  cvt_f32_f16<<<2048, 256, 0, stream>>>(in_v, vh, N_ * D_);

  // 2) transpose-convert weights (column-major B operands)
  tr_cvt<<<1024, 256, 0, stream>>>(in_wk, wk_t, D_, ED_, 1);
  tr_cvt<<<1024, 256, 0, stream>>>(in_wv, wv_t, D_, ED_, 1);
  tr_cvt<<<4096, 256, 0, stream>>>(in_wm, wmap_t, D_, ED_, E_);
  tr_cvt<<<4096, 256, 0, stream>>>(in_wr, wr_t, EED_, D_, 1);
  rpe_pad<<<160, 256, 0, stream>>>(in_rpe, rpet);

  // 3) gating + top-k (f32 for accuracy)
  gate_topk<<<N_, 128, 0, stream>>>(in_q, in_wg, gates, tidx);

  // 4) dense expert-map GEMM (2048x1024 @ 1024x3072), WMMA f16
  gemm_allmap<<<dim3(EED_ / 64, N_ / 64), 128, 0, stream>>>(xh, wmap_t, allmap);

  // 5) K/V projections (scaled K row-major, V transposed)
  gemm_proj<<<dim3(ED_ / 64, N_ / 64), 128, 0, stream>>>(kh, wk_t, khp, 0);
  gemm_proj<<<dim3(ED_ / 64, N_ / 64), 128, 0, stream>>>(vh, wv_t, vtb, 1);

  // 6) gather selected experts into scaled Q
  q_gather<<<2048, 256, 0, stream>>>(allmap, tidx, qh);

  // 7) zero z, then flash attention scatters gated outputs into expert slots
  zero_f16_v<<<1536, 256, 0, stream>>>(zbuf, (N_ * EED_) / 8);
  attn_kernel<<<dim3(T_ / 16, B_ * TK_ * H_), 32, 0, stream>>>(qh, khp, vtb, rpet,
                                                               gates, tidx, zbuf);

  // 8) final reduce GEMM (2048x3072 @ 3072x1024) -> f32 output
  gemm_out<<<dim3(D_ / 64, N_ / 64), 128, 0, stream>>>(zbuf, wr_t, out);

  (void)in_sizes; (void)n_in; (void)out_size; (void)ws_size;
}
